// DCRNNModel_pann_classification_40965398069647
// MI455X (gfx1250) — matrix operand order repaired and checked by reading, not compiled, not run on hardware
//
#include <hip/hip_runtime.h>
#include <hip/hip_bf16.h>

typedef __attribute__((ext_vector_type(16))) __bf16 v16bf;
typedef __attribute__((ext_vector_type(8)))  float  v8f;
typedef unsigned short ushort_t;

#define BATCH 64
#define TSTEPS 96
#define INW 128            // F+U = U+U = 128 for both layers
#define MTERMS 5           // 2K+1
#define XCOLS (INW*MTERMS) // 640, planar: j' = m*128 + f

__device__ __forceinline__ __bf16 f2bf(float x) { return (__bf16)x; }
__device__ __forceinline__ float bf2f(__bf16 h) { return (float)h; }

// ---- fragment loaders (all contiguous 16B accesses) ------------------------
__device__ __forceinline__ v16bf load_a_lds(const ushort_t* s, int row, int kt, int lane, int lda) {
    int cbase = kt * 32 + ((lane & 16) ? 16 : 0);
    const uint4* p = (const uint4*)(s + row * lda + cbase);
    union { uint4 q[2]; v16bf v; } u;
    u.q[0] = p[0];
    u.q[1] = p[1];
    return u.v;
}
__device__ __forceinline__ v16bf load_b_lds(const ushort_t* s, int rowBase, int kt, int lane) {
    int k0 = kt * 32 + ((lane & 16) ? 16 : 0);
    const uint4* p = (const uint4*)(s + rowBase + k0);
    union { uint4 q[2]; v16bf v; } u;
    u.q[0] = p[0]; u.q[1] = p[1];
    return u.v;
}
__device__ __forceinline__ v16bf load_frag_g(const ushort_t* g, int fragIdx, int lane) {
    const uint4* p = (const uint4*)(g + ((size_t)fragIdx * 32 + lane) * 16);
    union { uint4 q[2]; v16bf v; } u;
    u.q[0] = p[0]; u.q[1] = p[1];
    return u.v;
}

// ---- CDNA5 async global->LDS copy (ASYNCcnt) -------------------------------
__device__ __forceinline__ void async_copy_b128(const float* g, float* lds_generic) {
    unsigned ldsaddr = (unsigned)(size_t)(const void*)lds_generic; // low 32b = LDS offset
    asm volatile("global_load_async_to_lds_b128 %0, %1, off"
                 :: "v"(ldsaddr), "v"(g) : "memory");
}
__device__ __forceinline__ void wait_async0() {
    asm volatile("s_wait_asynccnt 0" ::: "memory");
}

// ---- GEMM1 (transposed): Y^T = x^T @ S^T, Y^T is 128(feat) x 64(node) ------
template<bool CHEB>
__device__ __forceinline__ void diffuse_gemm(ushort_t* sxcat, const ushort_t* xTin,
                                             ushort_t* x1T, const ushort_t* Sswz,
                                             int termOut, int wave, int lane) {
    const int nt = wave & 3, q = wave >> 2;    // nt fixed per wave; mt = q + 2*tl
    v8f acc[4] = {};
#pragma unroll
    for (int kt = 0; kt < 2; ++kt) {           // K = 64 (node dim)
        v16bf b = load_frag_g(Sswz, kt * 4 + nt, lane);
#pragma unroll
        for (int tl = 0; tl < 4; ++tl) {
            int mt = q + 2 * tl;
            v16bf a = load_a_lds(xTin, mt * 16 + (lane & 15), kt, lane, 64);
            acc[tl] = __builtin_amdgcn_wmma_f32_16x16x32_bf16(false, a, false, b,
                                                              (short)0, acc[tl], false, false);
        }
    }
#pragma unroll
    for (int tl = 0; tl < 4; ++tl) {
        int mt = q + 2 * tl;
        int Mb = mt * 16 + ((lane & 16) ? 8 : 0);   // feature base (8 consecutive)
        int N  = nt * 16 + (lane & 15);             // node
        float v[8];
#pragma unroll
        for (int j = 0; j < 8; ++j) v[j] = acc[tl][j];
        if (CHEB) {
            union { uint4 q4; __bf16 h[8]; } x0;
            x0.q4 = *(const uint4*)(sxcat + N * XCOLS + Mb);   // term 0, packed
#pragma unroll
            for (int j = 0; j < 8; ++j) v[j] = 2.0f * v[j] - bf2f(x0.h[j]);
        }
        union { uint4 q4; __bf16 h[8]; } pk;
#pragma unroll
        for (int j = 0; j < 8; ++j) pk.h[j] = f2bf(v[j]);
        *(uint4*)(sxcat + N * XCOLS + termOut * 128 + Mb) = pk.q4;  // packed b128
        if (!CHEB) {
#pragma unroll
            for (int j = 0; j < 8; ++j)
                *(__bf16*)(x1T + (Mb + j) * 64 + N) = pk.h[j];
        }
    }
}

// ---- One DCGRU half-step (gate or candidate), all state in LDS -------------
template<bool GATE, int LAYER>
__device__ __forceinline__ void half_step(
    ushort_t* sxcat, ushort_t* sxT0, ushort_t* sx1T,
    float* sh,                 // hidden state of this layer (64x64, LDS)
    float* sru,                // gate output r|u (64x128, LDS)
    float* sx0,                // inter-layer x buffer (64x64, LDS)
    const float* xsrc,         // f32 x source in LDS (sxin for L0, sx0 for L1)
    const ushort_t* sswzB, const ushort_t* wswzA, const float* bias,
    int tid, float* lastOut)   // global last_h slice for this batch, or nullptr
{
    const int lane = tid & 31, wave = tid >> 5;

    // Phase 0: build cat = [x, (GATE ? h : r*h)] into xcat block0 (packed) + cat^T
    for (int u = tid; u < 512; u += 256) {      // (node, 16-feature block)
        int n = u >> 3, fb = u & 7, f0 = fb * 16;
        float vals[16];
        if (fb < 4) {
#pragma unroll
            for (int i = 0; i < 16; i += 4) {
                float4 q = *(const float4*)(xsrc + n * 64 + f0 + i);
                vals[i] = q.x; vals[i+1] = q.y; vals[i+2] = q.z; vals[i+3] = q.w;
            }
        } else {
            int fp0 = f0 - 64;
#pragma unroll
            for (int i = 0; i < 16; i += 4) {
                float4 hq = *(const float4*)(sh + n * 64 + fp0 + i);
                if (!GATE) {
                    float4 rq = *(const float4*)(sru + n * 128 + fp0 + i);
                    hq.x *= rq.x; hq.y *= rq.y; hq.z *= rq.z; hq.w *= rq.w;
                }
                vals[i] = hq.x; vals[i+1] = hq.y; vals[i+2] = hq.z; vals[i+3] = hq.w;
            }
        }
        union { __bf16 h[16]; uint4 q[2]; } pk;
#pragma unroll
        for (int e = 0; e < 16; ++e) pk.h[e] = f2bf(vals[e]);
        uint4* dst = (uint4*)(sxcat + n * XCOLS + f0);
        dst[0] = pk.q[0]; dst[1] = pk.q[1];
#pragma unroll
        for (int e = 0; e < 16; ++e)
            *(__bf16*)(sxT0 + (f0 + e) * 64 + n) = pk.h[e];
    }
    __syncthreads();

    // Diffusion: terms 1..4 (planar blocks)
    diffuse_gemm<false>(sxcat, sxT0, sx1T, sswzB,        1, wave, lane); __syncthreads();
    diffuse_gemm<true >(sxcat, sx1T, sx1T, sswzB,        2, wave, lane); __syncthreads();
    diffuse_gemm<false>(sxcat, sxT0, sx1T, sswzB + 4096, 3, wave, lane); __syncthreads();
    diffuse_gemm<true >(sxcat, sx1T, sx1T, sswzB + 4096, 4, wave, lane); __syncthreads();

    // GEMM2 (transposed): G^T = W^T @ xcat^T, B fragment hoisted per kt
    constexpr int MTN  = GATE ? 8 : 4;    // Nout/16
    constexpr int NACC = MTN / 2;
    const int nt = wave & 3, q = wave >> 2;
    v8f acc[NACC] = {};
    const int nodeRow = (nt * 16 + (lane & 15)) * XCOLS;
    for (int kt = 0; kt < 20; ++kt) {     // K = 640
        v16bf bfr = load_b_lds(sxcat, nodeRow, kt, lane);
#pragma unroll
        for (int i = 0; i < NACC; ++i) {
            int mt = q + 2 * i;
            v16bf a = load_frag_g(wswzA, mt * 20 + kt, lane);
            acc[i] = __builtin_amdgcn_wmma_f32_16x16x32_bf16(false, a, false, bfr,
                                                             (short)0, acc[i], false, false);
        }
    }
#pragma unroll
    for (int i = 0; i < NACC; ++i) {
        int mt = q + 2 * i;
        int Mb = mt * 16 + ((lane & 16) ? 8 : 0);   // out-column base (8 consecutive)
        int N  = nt * 16 + (lane & 15);             // node
        float4 bb0 = *(const float4*)(bias + Mb);
        float4 bb1 = *(const float4*)(bias + Mb + 4);
        float g[8] = { acc[i][0]+bb0.x, acc[i][1]+bb0.y, acc[i][2]+bb0.z, acc[i][3]+bb0.w,
                       acc[i][4]+bb1.x, acc[i][5]+bb1.y, acc[i][6]+bb1.z, acc[i][7]+bb1.w };
        if (GATE) {
            float4 o0, o1;
            o0.x = 1.0f/(1.0f+expf(-g[0])); o0.y = 1.0f/(1.0f+expf(-g[1]));
            o0.z = 1.0f/(1.0f+expf(-g[2])); o0.w = 1.0f/(1.0f+expf(-g[3]));
            o1.x = 1.0f/(1.0f+expf(-g[4])); o1.y = 1.0f/(1.0f+expf(-g[5]));
            o1.z = 1.0f/(1.0f+expf(-g[6])); o1.w = 1.0f/(1.0f+expf(-g[7]));
            *(float4*)(sru + N * 128 + Mb)     = o0;
            *(float4*)(sru + N * 128 + Mb + 4) = o1;
        } else {
            float4 u0 = *(const float4*)(sru + N * 128 + 64 + Mb);
            float4 u1 = *(const float4*)(sru + N * 128 + 64 + Mb + 4);
            float4 h0v = *(const float4*)(sh + N * 64 + Mb);
            float4 h1v = *(const float4*)(sh + N * 64 + Mb + 4);
            float4 n0, n1;
            n0.x = u0.x*h0v.x + (1.0f-u0.x)*tanhf(g[0]);
            n0.y = u0.y*h0v.y + (1.0f-u0.y)*tanhf(g[1]);
            n0.z = u0.z*h0v.z + (1.0f-u0.z)*tanhf(g[2]);
            n0.w = u0.w*h0v.w + (1.0f-u0.w)*tanhf(g[3]);
            n1.x = u1.x*h1v.x + (1.0f-u1.x)*tanhf(g[4]);
            n1.y = u1.y*h1v.y + (1.0f-u1.y)*tanhf(g[5]);
            n1.z = u1.z*h1v.z + (1.0f-u1.z)*tanhf(g[6]);
            n1.w = u1.w*h1v.w + (1.0f-u1.w)*tanhf(g[7]);
            *(float4*)(sh + N * 64 + Mb)     = n0;
            *(float4*)(sh + N * 64 + Mb + 4) = n1;
            if (LAYER == 0) {
                *(float4*)(sx0 + N * 64 + Mb)     = n0;   // feeds layer 1
                *(float4*)(sx0 + N * 64 + Mb + 4) = n1;
            } else if (lastOut) {
                *(float4*)(lastOut + N * 64 + Mb)     = n0;
                *(float4*)(lastOut + N * 64 + Mb + 4) = n1;
            }
        }
    }
    __syncthreads();
}

// ---- persistent encoder: one block per batch, full 96-step recurrence ------
__global__ __launch_bounds__(256) void encoder_kernel(
    const float* __restrict__ input_seq, const int* __restrict__ seq_len,
    const ushort_t* __restrict__ sswzB,
    const ushort_t* __restrict__ wg0s, const ushort_t* __restrict__ wc0s,
    const ushort_t* __restrict__ wg1s, const ushort_t* __restrict__ wc1s,
    const float* __restrict__ bg0, const float* __restrict__ bc0,
    const float* __restrict__ bg1, const float* __restrict__ bc1,
    float* __restrict__ lasth)
{
    __shared__ ushort_t sxcat[64 * XCOLS]; // 80 KB planar xcat
    __shared__ ushort_t sxT0[128 * 64];    // 16 KB cat^T
    __shared__ ushort_t sx1T[128 * 64];    // 16 KB S@x transposed
    __shared__ float    sh0[64 * 64];      // 16 KB layer-0 hidden
    __shared__ float    sh1[64 * 64];      // 16 KB layer-1 hidden
    __shared__ float    sru[64 * 128];     // 32 KB r|u
    __shared__ float    sx0[64 * 64];      // 16 KB inter-layer buffer
    __shared__ float    sxin[2][64 * 64];  // 32 KB double-buffered input tile

    const int b = blockIdx.x;
    const int tid = threadIdx.x;
    const int tlast = seq_len[b] - 1;
    float* lastSlice = lasth + (size_t)b * 4096;
    const float* gbase = input_seq + ((size_t)b * TSTEPS) * 4096;

    for (int i = tid; i < 4096; i += 256) { sh0[i] = 0.f; sh1[i] = 0.f; }
    // synchronous copy of x(0)
    for (int i = tid; i < 1024; i += 256)
        ((float4*)sxin[0])[i] = ((const float4*)gbase)[i];
    __syncthreads();

#pragma unroll 1
    for (int t = 0; t < TSTEPS; ++t) {
        // async-prefetch x(t+1) into the other LDS buffer (ASYNCcnt-tracked)
        if (t + 1 < TSTEPS) {
            const float* gnext = gbase + (size_t)(t + 1) * 4096;
            float* dstbuf = sxin[(t + 1) & 1];
#pragma unroll
            for (int k = 0; k < 4; ++k) {
                int elem = tid * 4 + k * 1024;
                async_copy_b128(gnext + elem, dstbuf + elem);
            }
        }
        const float* xin = sxin[t & 1];
        half_step<true , 0>(sxcat, sxT0, sx1T, sh0, sru, sx0, xin, sswzB, wg0s, bg0, tid, nullptr);
        half_step<false, 0>(sxcat, sxT0, sx1T, sh0, sru, sx0, xin, sswzB, wc0s, bc0, tid, nullptr);
        half_step<true , 1>(sxcat, sxT0, sx1T, sh1, sru, sx0, sx0, sswzB, wg1s, bg1, tid, nullptr);
        half_step<false, 1>(sxcat, sxT0, sx1T, sh1, sru, sx0, sx0, sswzB, wc1s, bc1, tid,
                            (t == tlast) ? lastSlice : nullptr);
        wait_async0();        // x(t+1) resident in LDS
        __syncthreads();
    }
}

// ---- prep: swizzle supports (B-layout of S^T) and weights (A-layout of W^T) -
__device__ void swzS_B(const float* S, ushort_t* dst, int gt, int stride) {
    for (int idx = gt; idx < 4096; idx += stride) {
        int e = idx & 15, lane = (idx >> 4) & 31, fn = idx >> 9;
        int nt = fn & 3, kt = fn >> 2;
        int k = kt * 32 + ((lane & 16) ? 16 : 0) + e;
        int mcol = nt * 16 + (lane & 15);
        *(__bf16*)(dst + idx) = f2bf(S[mcol * 64 + k]);     // S^T[k][m] = S[m][k]
    }
}
__device__ void swzW_A(const float* W, int ncols, ushort_t* dst, int gt, int stride) {
    int mtn = ncols / 16;
    int total = mtn * 20 * 512;
    for (int idx = gt; idx < total; idx += stride) {
        int e = idx & 15, lane = (idx >> 4) & 31, fn = idx >> 9;
        int kt = fn % 20, mt = fn / 20;
        int row = mt * 16 + (lane & 15);
        int k = kt * 32 + ((lane & 16) ? 16 : 0) + e;
        int m = k >> 7, f = k & 127;                        // planar j' -> (m,f)
        int jorig = f * 5 + m;
        *(__bf16*)(dst + idx) = f2bf(W[(size_t)jorig * ncols + row]);
    }
}
__global__ void prep_kernel(const float* S0, const float* S1,
                            const float* Wg0, const float* Wc0,
                            const float* Wg1, const float* Wc1,
                            ushort_t* wg0s, ushort_t* wc0s,
                            ushort_t* wg1s, ushort_t* wc1s,
                            ushort_t* sswz)
{
    int gt = blockIdx.x * blockDim.x + threadIdx.x;
    int stride = gridDim.x * blockDim.x;
    swzW_A(Wg0, 128, wg0s, gt, stride);
    swzW_A(Wc0,  64, wc0s, gt, stride);
    swzW_A(Wg1, 128, wg1s, gt, stride);
    swzW_A(Wc1,  64, wc1s, gt, stride);
    swzS_B(S0, sswz,        gt, stride);
    swzS_B(S1, sswz + 4096, gt, stride);
}

// ---- head: relu, two small GEMMs, max over nodes ---------------------------
__global__ void final_kernel(const float* __restrict__ last_h,
                             const float* fc_w, const float* fc_b,
                             const float* id_w, const float* id_b,
                             float* __restrict__ out)
{
    __shared__ float hl[64 * 64];
    int b = blockIdx.x, tid = threadIdx.x;
    for (int i = tid; i < 4096; i += blockDim.x) {
        float v = last_h[(size_t)b * 4096 + i];
        hl[i] = v > 0.f ? v : 0.f;
    }
    __syncthreads();
    for (int o = tid; o < 54; o += blockDim.x) {
        const float* W; float bias; int C, c;
        if (o < 4) { W = fc_w; C = 4;  c = o;     bias = fc_b[c]; }
        else       { W = id_w; C = 50; c = o - 4; bias = id_b[c]; }
        float best = -3.4e38f;
#pragma unroll 1
        for (int n = 0; n < 64; ++n) {
            float s = 0.f;
#pragma unroll 1
            for (int u = 0; u < 64; ++u) s += hl[n * 64 + u] * W[u * C + c];
            best = fmaxf(best, s);
        }
        best += bias;
        if (o < 4) out[b * 4 + c] = best;
        else       out[256 + b * 50 + c] = best;
    }
}

extern "C" void kernel_launch(void* const* d_in, const int* in_sizes, int n_in,
                              void* d_out, int out_size, void* d_ws, size_t ws_size,
                              hipStream_t stream) {
    const float* input_seq = (const float*)d_in[0];
    const int*   seq_len   = (const int*)  d_in[1];
    const float* S0  = (const float*)d_in[2];
    const float* S1  = (const float*)d_in[3];
    const float* Wg0 = (const float*)d_in[4];
    const float* bg0 = (const float*)d_in[5];
    const float* Wc0 = (const float*)d_in[6];
    const float* bc0 = (const float*)d_in[7];
    const float* Wg1 = (const float*)d_in[8];
    const float* bg1 = (const float*)d_in[9];
    const float* Wc1 = (const float*)d_in[10];
    const float* bc1 = (const float*)d_in[11];
    const float* fc_w = (const float*)d_in[12];
    const float* fc_b = (const float*)d_in[13];
    const float* id_w = (const float*)d_in[14];
    const float* id_b = (const float*)d_in[15];
    float* out = (float*)d_out;

    char* ws = (char*)d_ws;
    size_t off = 0;
    float* lasth = (float*)(ws + off); off += (size_t)BATCH * 64 * 64 * 4;   // 1 MB
    ushort_t* wg0s = (ushort_t*)(ws + off); off += 81920u * 2;
    ushort_t* wc0s = (ushort_t*)(ws + off); off += 40960u * 2;
    ushort_t* wg1s = (ushort_t*)(ws + off); off += 81920u * 2;
    ushort_t* wc1s = (ushort_t*)(ws + off); off += 40960u * 2;
    ushort_t* sswz = (ushort_t*)(ws + off); off += 8192u * 2;

    prep_kernel<<<128, 256, 0, stream>>>(S0, S1, Wg0, Wc0, Wg1, Wc1,
                                         wg0s, wc0s, wg1s, wc1s, sswz);

    encoder_kernel<<<BATCH, 256, 0, stream>>>(input_seq, seq_len, sswz,
                                              wg0s, wc0s, wg1s, wc1s,
                                              bg0, bc0, bg1, bc1, lasth);

    final_kernel<<<BATCH, 64, 0, stream>>>(lasth, fc_w, fc_b, id_w, id_b, out);
}
